// MultiHeadAttention_44908178047033
// MI455X (gfx1250) — matrix operand order repaired and checked by reading, not compiled, not run on hardware
//
#include <hip/hip_runtime.h>

// ---------------------------------------------------------------------------
// T5-style MHA for MI455X (gfx1250, wave32, WMMA).
// All GEMMs: bf16 inputs, fp32 accumulate via v_wmma_f32_16x16x32_bf16.
// Fragment loads are batched ahead of the WMMA chains so ds_load_b128s issue
// as clauses and WMMAs run back-to-back; global_prefetch_b8 overlaps the next
// tile fetch with current-tile math.
// ---------------------------------------------------------------------------

typedef unsigned short u16t;
typedef __attribute__((ext_vector_type(16))) __bf16        v16bf;
typedef __attribute__((ext_vector_type(8)))  float         v8f;
typedef __attribute__((ext_vector_type(4)))  unsigned int  v4u;

#define BB   2
#define LL   2048
#define DD   768
#define HH   12
#define DKK  64
#define MM   (BB * LL)      // 4096 rows

union FragBF { v16bf bf; v4u u[2]; };

__device__ __forceinline__ u16t f2bf(float f) {
  unsigned u = __float_as_uint(f);
  u += 0x7FFFu + ((u >> 16) & 1u);         // round-to-nearest-even
  return (u16t)(u >> 16);
}

// A-matrix fragment (16x32 bf16), source row-major [row][k], ld in elements.
// ISA 16-bit A layout: lane m=l%16 holds row m; K = 2i+p + (i>=4?8:0) + 8*half.
__device__ __forceinline__ FragBF load_fragA(const u16t* p, int ld, int r0, int k0) {
  const int lane = threadIdx.x & 31;
  const int m    = lane & 15;
  const int hlf  = lane >> 4;
  const u16t* row = p + (size_t)(r0 + m) * ld + k0 + 8 * hlf;
  FragBF f;
  f.u[0] = *(const v4u*)(row);        // K = k0+8h .. +7
  f.u[1] = *(const v4u*)(row + 16);   // K = k0+16+8h .. +7
  return f;
}

// B-matrix fragment (32x16 bf16) from a TRANSPOSED tile [N][K] (ld elements).
// ISA B layout: lane n=l%16 holds col n; K = 2i+p + (i>=4?8:0) + 8*half.
__device__ __forceinline__ FragBF load_fragB(const u16t* p, int ld, int n0, int k0) {
  const int lane = threadIdx.x & 31;
  const int n    = lane & 15;
  const int hlf  = lane >> 4;
  const u16t* row = p + (size_t)(n0 + n) * ld + k0 + 8 * hlf;
  FragBF f;
  f.u[0] = *(const v4u*)(row);
  f.u[1] = *(const v4u*)(row + 16);
  return f;
}

__device__ __forceinline__ v8f wmma_bf16(const FragBF& a, const FragBF& b, v8f c) {
  return __builtin_amdgcn_wmma_f32_16x16x32_bf16(false, a.bf, false, b.bf,
                                                 (short)0, c, false, false);
}

// ---------------------------------------------------------------------------
// fp32 -> bf16 convert
// ---------------------------------------------------------------------------
__global__ __launch_bounds__(256) void cvt_kernel(const float* __restrict__ src,
                                                  u16t* __restrict__ dst, int n) {
  int i = blockIdx.x * 256 + threadIdx.x;
  if (i < n) dst[i] = f2bf(src[i]);
}

// ---------------------------------------------------------------------------
// T5 relative-position bias table: biasTab[h][rel+2047], rel = k - q.
// bidirectional, num_buckets=32 (16/side), max_exact=8, max_distance=128.
// ---------------------------------------------------------------------------
__global__ __launch_bounds__(256) void bias_table_kernel(const float* __restrict__ rel_emb,
                                                         float* __restrict__ biasTab) {
  int i = blockIdx.x * 256 + threadIdx.x;   // 0 .. 4094
  if (i >= 2 * LL - 1) return;
  int rel = i - (LL - 1);                   // k - q
  int n   = -rel;
  int ret = (n < 0) ? 16 : 0;
  int an  = (n < 0) ? -n : n;
  int bucket;
  if (an < 8) {
    bucket = an;
  } else {
    float v = __logf((float)an * 0.125f) * (1.0f / __logf(16.0f)) * 8.0f;
    int vi = 8 + (int)v;
    bucket = vi < 15 ? vi : 15;
  }
  bucket += ret;
  for (int h = 0; h < HH; ++h)
    biasTab[h * (2 * LL - 1) + i] = rel_emb[bucket * HH + h];
}

// ---------------------------------------------------------------------------
// GEMM: Y = X[4096x768] * W[768x768], bf16 in, fp32 acc.
// mode 0: write bf16 to [B,H,L,DK];  mode 1: write fp32 flat [4096][768].
// Block 256 (8 waves), tile 128x128, k-step 64 (two 32-k WMMA chunks).
// ---------------------------------------------------------------------------
#define GTK 64
#define GLD 72   // 64 + 8 pad (elements), row stride 144 B (16B aligned)
__global__ __launch_bounds__(256) void gemm_kernel(const u16t* __restrict__ X,
                                                   const u16t* __restrict__ W,
                                                   void* __restrict__ Yout, int mode) {
  __shared__ __attribute__((aligned(16))) u16t lx [128 * GLD];   // X tile [m][k]
  __shared__ __attribute__((aligned(16))) u16t lwt[128 * GLD];   // W tile transposed [n][k]

  const int m0   = (int)blockIdx.x * 128;
  const int n0   = (int)blockIdx.y * 128;
  const int tid  = threadIdx.x;
  const int wave = tid >> 5;
  const int lane = tid & 31;
  const int hlf  = lane >> 4;
  const int nl   = lane & 15;

  v8f acc[8];
  for (int t = 0; t < 8; ++t)
    for (int j = 0; j < 8; ++j) acc[t][j] = 0.0f;

  for (int k0 = 0; k0 < DD; k0 += GTK) {
    __syncthreads();
    // X tile: 128x64, vectorized 8-element copies (coalesced, 16B aligned)
    for (int e = tid; e < 128 * GTK / 8; e += 256) {
      int m = e >> 3, kk = (e & 7) * 8;
      *(v4u*)&lx[m * GLD + kk] = *(const v4u*)&X[(size_t)(m0 + m) * DD + k0 + kk];
    }
    // W tile transposed: 64x128 -> lwt[n][k] (coalesced global reads)
    for (int e = tid; e < 128 * GTK; e += 256) {
      int k = e >> 7, n = e & 127;
      lwt[n * GLD + k] = W[(size_t)(k0 + k) * DD + n0 + n];
    }
    // prefetch next tiles into cache while this tile computes
    if (k0 + GTK < DD) {
      __builtin_prefetch(&X[(size_t)(m0 + (tid >> 1)) * DD + k0 + GTK + (tid & 1) * 32], 0, 0);
      __builtin_prefetch(&W[(size_t)(k0 + GTK + (tid >> 3)) * DD + n0 + (tid & 7) * 16], 0, 0);
    }
    __syncthreads();

    for (int kc = 0; kc < GTK; kc += 32) {
      FragBF a = load_fragA(lx, GLD, wave * 16, kc);
      FragBF b[8];
      for (int nt = 0; nt < 8; ++nt) b[nt] = load_fragB(lwt, GLD, nt * 16, kc);
      for (int nt = 0; nt < 8; ++nt) acc[nt] = wmma_bf16(a, b[nt], acc[nt]);
    }
  }

  // C/D layout: row = j + 8*half, col = lane%16
  const int mrow0 = m0 + wave * 16;
  for (int nt = 0; nt < 8; ++nt) {
    int n = n0 + nt * 16 + nl;
    for (int j = 0; j < 8; ++j) {
      int m = mrow0 + j + 8 * hlf;
      float v = acc[nt][j];
      if (mode == 0) {
        int b = m >> 11, l = m & (LL - 1);
        int h = n >> 6,  dk = n & 63;
        ((u16t*)Yout)[((size_t)((b * HH + h) * LL + l)) * DKK + dk] = f2bf(v);
      } else {
        ((float*)Yout)[(size_t)m * DD + n] = v;
      }
    }
  }
}

// ---------------------------------------------------------------------------
// Flash attention: one block per (b, h, 128-row q tile). 8 waves; each wave
// owns 16 q rows. K tiles of 64. Online softmax in fp32, P via per-wave LDS
// scratch re-layout (C-layout -> A-fragment) for the P*V WMMA.
// ---------------------------------------------------------------------------
#define QBLK 128
#define KBLK 64
__global__ __launch_bounds__(256) void attn_kernel(const u16t* __restrict__ Qg,
                                                   const u16t* __restrict__ Kg,
                                                   const u16t* __restrict__ Vg,
                                                   const int*  __restrict__ maskg,
                                                   const float* __restrict__ biasTab,
                                                   u16t* __restrict__ Og) {
  __shared__ __attribute__((aligned(16))) u16t  lk [KBLK * 72];        // K tile [kpos][dk]  (= B^T for scores)
  __shared__ __attribute__((aligned(16))) u16t  lvt[DKK * (KBLK + 8)]; // V^T tile [dk][kpos] (= B^T for P*V)
  __shared__ __attribute__((aligned(16))) float lbias[4096];           // bias by rel for this head
  __shared__ __attribute__((aligned(16))) u16t  lp [8 * 16 * KBLK];    // per-wave P scratch [16][KBLK]

  const int nQ  = LL / QBLK;                    // 16
  const int bid = blockIdx.x;
  const int b   = bid / (HH * nQ);
  const int h   = (bid / nQ) % HH;
  const int q0  = (bid % nQ) * QBLK;
  const int tid = threadIdx.x, wave = tid >> 5, lane = tid & 31;
  const int hlf = lane >> 4, nl = lane & 15;

  for (int i = tid; i < 2 * LL - 1; i += 256) lbias[i] = biasTab[h * (2 * LL - 1) + i];

  const size_t head = ((size_t)(b * HH + h)) * LL * DKK;
  const u16t* Qh = Qg + head;
  const u16t* Kh = Kg + head;
  const u16t* Vh = Vg + head;

  // Q fragments for this wave's 16 rows (kept in registers for whole kernel)
  FragBF qf0 = load_fragA(Qh, DKK, q0 + wave * 16, 0);
  FragBF qf1 = load_fragA(Qh, DKK, q0 + wave * 16, 32);

  float mrun[8], lrun[8];
  v8f oacc[4];
  for (int j = 0; j < 8; ++j) { mrun[j] = -1e30f; lrun[j] = 0.0f; }
  for (int d = 0; d < 4; ++d)
    for (int j = 0; j < 8; ++j) oacc[d][j] = 0.0f;

  const float scale = 0.125f;   // 1/sqrt(64)
  u16t* pw = &lp[wave * 16 * KBLK];

  for (int kt = 0; kt < LL / KBLK; ++kt) {
    const int kbase = kt * KBLK;
    __syncthreads();
    // stage K tile (row-major is already B^T layout), vectorized
    for (int e = tid; e < KBLK * DKK / 8; e += 256) {
      int kp = e >> 3, kk = (e & 7) * 8;
      *(v4u*)&lk[kp * 72 + kk] = *(const v4u*)&Kh[(size_t)(kbase + kp) * DKK + kk];
    }
    // stage V transposed: lvt[dk][kpos]
    for (int e = tid; e < KBLK * DKK; e += 256) {
      int kp = e >> 6, dk = e & 63;
      lvt[dk * (KBLK + 8) + kp] = Vh[(size_t)(kbase + kp) * DKK + dk];
    }
    // prefetch next K/V tiles while this one computes
    if (kbase + KBLK < LL) {
      __builtin_prefetch(&Kh[(size_t)(kbase + KBLK + (tid >> 2)) * DKK + (tid & 3) * 16], 0, 0);
      __builtin_prefetch(&Vh[(size_t)(kbase + KBLK + (tid >> 2)) * DKK + (tid & 3) * 16], 0, 0);
    }
    __syncthreads();

    // S = Q K^T: batch all K fragments, then chain 8 WMMAs
    FragBF kf[4][2];
    for (int nt = 0; nt < 4; ++nt) {
      kf[nt][0] = load_fragB(lk, 72, nt * 16, 0);
      kf[nt][1] = load_fragB(lk, 72, nt * 16, 32);
    }
    v8f s[4];
    for (int nt = 0; nt < 4; ++nt) {
      v8f a;
      for (int j = 0; j < 8; ++j) a[j] = 0.0f;
      a = wmma_bf16(qf0, kf[nt][0], a);
      a = wmma_bf16(qf1, kf[nt][1], a);
      s[nt] = a;
    }

    // scale + T5 bias + padding mask
    for (int nt = 0; nt < 4; ++nt) {
      int kcol = kbase + nt * 16 + nl;
      int mk = maskg[b * LL + kcol];
      for (int j = 0; j < 8; ++j) {
        int qrow = q0 + wave * 16 + j + 8 * hlf;
        float v = s[nt][j] * scale + lbias[kcol - qrow + (LL - 1)];
        s[nt][j] = mk ? v : -1e30f;
      }
    }

    // online softmax (rows live in 16-lane groups -> xor-shuffle reductions)
    float pfac[8];
    for (int j = 0; j < 8; ++j) {
      float mx = s[0][j];
      for (int nt = 1; nt < 4; ++nt) mx = fmaxf(mx, s[nt][j]);
      for (int d = 1; d < 16; d <<= 1) mx = fmaxf(mx, __shfl_xor(mx, d, 32));
      float mnew = fmaxf(mrun[j], mx);
      pfac[j] = __expf(mrun[j] - mnew);
      mrun[j] = mnew;
      lrun[j] *= pfac[j];
      float rs = 0.0f;
      for (int nt = 0; nt < 4; ++nt) {
        float p = __expf(s[nt][j] - mnew);
        s[nt][j] = p;
        rs += p;
      }
      for (int d = 1; d < 16; d <<= 1) rs += __shfl_xor(rs, d, 32);
      lrun[j] += rs;
    }
    for (int d = 0; d < 4; ++d)
      for (int j = 0; j < 8; ++j) oacc[d][j] *= pfac[j];

    // P (C-layout) -> per-wave LDS scratch as bf16 row-major [16][KBLK]
    for (int nt = 0; nt < 4; ++nt)
      for (int j = 0; j < 8; ++j)
        pw[(j + 8 * hlf) * KBLK + nt * 16 + nl] = f2bf(s[nt][j]);

    // O += P * V : batch P and V fragments per 32-k chunk, then chain WMMAs
    for (int kk = 0; kk < KBLK; kk += 32) {
      FragBF pf = load_fragA(pw, KBLK, 0, kk);
      FragBF vf[4];
      for (int d = 0; d < 4; ++d) vf[d] = load_fragB(lvt, KBLK + 8, d * 16, kk);
      for (int d = 0; d < 4; ++d) oacc[d] = wmma_bf16(pf, vf[d], oacc[d]);
    }
  }

  // finalize: O /= l, write bf16 [B*L][H*DK] for the output projection
  for (int d = 0; d < 4; ++d) {
    for (int j = 0; j < 8; ++j) {
      int qrow = q0 + wave * 16 + j + 8 * hlf;
      int dk = d * 16 + nl;
      float v = oacc[d][j] / lrun[j];
      Og[((size_t)(b * LL + qrow)) * DD + h * DKK + dk] = f2bf(v);
    }
  }
}

// ---------------------------------------------------------------------------
// Host-side orchestration (graph-capture safe: kernel launches only)
// ---------------------------------------------------------------------------
extern "C" void kernel_launch(void* const* d_in, const int* in_sizes, int n_in,
                              void* d_out, int out_size, void* d_ws, size_t ws_size,
                              hipStream_t stream) {
  const float* query     = (const float*)d_in[0];
  const float* key_value = (const float*)d_in[1];
  const int*   mask      = (const int*)  d_in[2];
  const float* Wq        = (const float*)d_in[3];
  const float* Wk        = (const float*)d_in[4];
  const float* Wv        = (const float*)d_in[5];
  const float* Wo        = (const float*)d_in[6];
  const float* rel_emb   = (const float*)d_in[7];

  char* ws = (char*)d_ws;
  size_t off = 0;
  auto alloc = [&](size_t bytes) -> void* {
    void* p = ws + off;
    off += (bytes + 255) & ~(size_t)255;
    return p;
  };

  u16t* q_bf   = (u16t*)alloc((size_t)MM * DD * 2);
  u16t* kv_bf  = (u16t*)alloc((size_t)MM * DD * 2);
  u16t* Wq_bf  = (u16t*)alloc((size_t)DD * DD * 2);
  u16t* Wk_bf  = (u16t*)alloc((size_t)DD * DD * 2);
  u16t* Wv_bf  = (u16t*)alloc((size_t)DD * DD * 2);
  u16t* Wo_bf  = (u16t*)alloc((size_t)DD * DD * 2);
  u16t* Qb     = (u16t*)alloc((size_t)MM * DD * 2);
  u16t* Kb     = (u16t*)alloc((size_t)MM * DD * 2);
  u16t* Vb     = (u16t*)alloc((size_t)MM * DD * 2);
  u16t* Obuf   = (u16t*)alloc((size_t)MM * DD * 2);
  float* biasT = (float*)alloc((size_t)HH * (2 * LL - 1) * 4);

  const int nAct = MM * DD;      // 3,145,728
  const int nW   = DD * DD;      // 589,824
  cvt_kernel<<<(nAct + 255) / 256, 256, 0, stream>>>(query,     q_bf,  nAct);
  cvt_kernel<<<(nAct + 255) / 256, 256, 0, stream>>>(key_value, kv_bf, nAct);
  cvt_kernel<<<(nW   + 255) / 256, 256, 0, stream>>>(Wq, Wq_bf, nW);
  cvt_kernel<<<(nW   + 255) / 256, 256, 0, stream>>>(Wk, Wk_bf, nW);
  cvt_kernel<<<(nW   + 255) / 256, 256, 0, stream>>>(Wv, Wv_bf, nW);
  cvt_kernel<<<(nW   + 255) / 256, 256, 0, stream>>>(Wo, Wo_bf, nW);

  bias_table_kernel<<<(2 * LL - 1 + 255) / 256, 256, 0, stream>>>(rel_emb, biasT);

  dim3 gg(MM / 128, DD / 128);   // 32 x 6
  gemm_kernel<<<gg, 256, 0, stream>>>(q_bf,  Wq_bf, (void*)Qb, 0);
  gemm_kernel<<<gg, 256, 0, stream>>>(kv_bf, Wk_bf, (void*)Kb, 0);
  gemm_kernel<<<gg, 256, 0, stream>>>(kv_bf, Wv_bf, (void*)Vb, 0);

  attn_kernel<<<BB * HH * (LL / QBLK), 256, 0, stream>>>(Qb, Kb, Vb, mask, biasT, Obuf);

  gemm_kernel<<<gg, 256, 0, stream>>>(Obuf, Wo_bf, d_out, 1);
}
